// MultiheadAttention_24833500905946
// MI455X (gfx1250) — matrix operand order repaired and checked by reading, not compiled, not run on hardware
//
#include <hip/hip_runtime.h>

// ---------------------------------------------------------------------------
// MHA forward for gfx1250 (MI455X): bf16 WMMA GEMMs (TDM-staged B tiles)
// + flash attention.
// ---------------------------------------------------------------------------

typedef __attribute__((ext_vector_type(16))) __bf16       v16bf;
typedef __attribute__((ext_vector_type(8)))  float        v8f;
typedef __attribute__((ext_vector_type(4)))  unsigned int v4u;
typedef __attribute__((ext_vector_type(4)))  unsigned int u32x4;
typedef __attribute__((ext_vector_type(8)))  int          i32x8;
typedef __attribute__((ext_vector_type(4)))  int          i32x4;

#define D_MODEL 1024
#define NHEADS  16
#define DKH     64
#define SEQ     2048
#define BATCH   4
#define MTOT    (BATCH * SEQ)   /* 8192 rows */
#define NK      (D_MODEL / 32)  /* 32 k-steps */
#define NEG_INF (-3.0e38f)

#if __has_builtin(__builtin_amdgcn_tensor_load_to_lds)
#define HAVE_TDM 1
#else
#define HAVE_TDM 0
#endif

// float -> bf16 (round-to-nearest-even)
static __device__ __forceinline__ unsigned short f2bf(float f) {
    unsigned int u = __builtin_bit_cast(unsigned int, f);
    u += 0x7fffu + ((u >> 16) & 1u);
    return (unsigned short)(u >> 16);
}

union FragU { v4u q[2]; v16bf v; };

// A fragment, 16x32 bf16 (row-major source, row = lane%16):
//   lanes 0-15 : K = k0+0..7  and k0+16..23
//   lanes 16-31: K = k0+8..15 and k0+24..31
static __device__ __forceinline__ v16bf load_a(const unsigned short* rowbase,
                                               int k0, int lane) {
    FragU u;
    const unsigned short* p = rowbase + k0 + ((lane & 16) ? 8 : 0);
    u.q[0] = *(const v4u*)(p);
    u.q[1] = *(const v4u*)(p + 16);
    return u.v;
}

// B fragment, 32x16 bf16 from K-contiguous storage (colbase = &BT[n][k0]):
//   lanes 0-15 : K = +0..7 and +16..23 ; lanes 16-31 : K = +8..15 and +24..31
static __device__ __forceinline__ v16bf load_b(const unsigned short* colbase,
                                               int lane) {
    FragU u;
    const unsigned short* p = colbase + ((lane & 16) ? 8 : 0);
    u.q[0] = *(const v4u*)(p);
    u.q[1] = *(const v4u*)(p + 16);
    return u.v;
}

static __device__ __forceinline__ v8f wmma_bf16(v16bf a, v16bf b, v8f c) {
    return __builtin_amdgcn_wmma_f32_16x16x32_bf16(
        false, a, false, b, (short)0, c, false, false);
}

// ---------------------------------------------------------------------------
// TDM helpers: DMA a 64-row x 32-elem bf16 tile (rows stride = 1024 elems)
// from global into LDS.  Descriptor per CDNA5 ISA.
// ---------------------------------------------------------------------------
template <int N>
static __device__ __forceinline__ void wait_tensorcnt() {
#if __has_builtin(__builtin_amdgcn_s_wait_tensorcnt)
    __builtin_amdgcn_s_wait_tensorcnt((short)N);
#else
    if (N == 0) asm volatile("s_wait_tensorcnt 0x0" ::: "memory");
    else        asm volatile("s_wait_tensorcnt 0x1" ::: "memory");
#endif
}

#if HAVE_TDM
static __device__ __forceinline__ void tdm_load_tile_64x32(
    const unsigned short* gsrc, unsigned lds_byte) {
    unsigned long long ga = (unsigned long long)(uintptr_t)gsrc;
    // group0: count=1 | lds_addr | global_addr(57b) | type=2
    u32x4 g0 = { 1u,
                 lds_byte,
                 (unsigned)(ga & 0xffffffffu),
                 ((unsigned)((ga >> 32) & 0x01ffffffu)) | 0x80000000u };
    // group1: data_size=1 (2B); tensor_dim0=tensor_dim1=2^20 (never OOB);
    //         tile_dim0=32 (contiguous k), tile_dim1=64 (rows),
    //         tensor_dim0_stride=1024; tile_dim2=0 -> 2D
    i32x8 g1 = { 0x00010000,           // workgroup_mask=0, data_size=1
                 0,                    // tensor_dim0[15:0] (2^20 -> 0)
                 0x10,                 // tensor_dim0[31:16]=0x10, dim1[15:0]=0
                 0x10 | (32 << 16),    // dim1[31:16]=0x10, tile_dim0=32
                 64,                   // tile_dim1=64, tile_dim2=0
                 D_MODEL,              // tensor_dim0_stride low32
                 0, 0 };               // stride0 hi, stride1 (unused)
    i32x4 z4 = { 0, 0, 0, 0 };
#if __clang_major__ >= 23
    i32x8 z8 = { 0, 0, 0, 0, 0, 0, 0, 0 };
    __builtin_amdgcn_tensor_load_to_lds(g0, g1, z4, z4, z8, 0);
#else
    __builtin_amdgcn_tensor_load_to_lds(g0, g1, z4, z4, 0);
#endif
}
#endif

// fallback: cooperative synchronous stage of the 64x32 tile (128 threads)
static __device__ __forceinline__ void stage_tile_fallback(
    const unsigned short* gsrc, unsigned short* lbuf, int tid) {
    const int idx = tid * 16;             // 16 elems (32B) per thread
    const int row = idx >> 5, col = idx & 31;
    const v4u* s = (const v4u*)(gsrc + (size_t)row * D_MODEL + col);
    v4u* d = (v4u*)(lbuf + idx);
    d[0] = s[0];
    d[1] = s[1];
}

// ---------------------------------------------------------------------------
// shared GEMM mainloop: block = 4 waves, block tile 128(M) x 64(N),
// each wave owns 32(M) x 64(N) = 8 accumulators.  B tile staged in LDS by TDM.
//   Abase0/1: A rows (m + lane%16) and (m + 16 + lane%16), K-major bf16
//   Wtile   : &BT[n0][0] (rows = output cols, 1024-elem stride)
// ---------------------------------------------------------------------------
static __device__ __forceinline__ void gemm_mainloop(
    const unsigned short* Abase0, const unsigned short* Abase1,
    const unsigned short* Wtile,
    unsigned short* lbuf0, unsigned short* lbuf1,
    int lane, int wv, int tid, v8f acc[2][4]) {

    const int n16 = lane & 15;
#if HAVE_TDM
    if (wv == 0) tdm_load_tile_64x32(Wtile, (unsigned)(uintptr_t)lbuf0);
#endif
    for (int i = 0; i < NK; ++i) {
        unsigned short* cur = (i & 1) ? lbuf1 : lbuf0;
        __syncthreads();   // prior readers done with buffer being refilled
#if HAVE_TDM
        if (wv == 0) {
            if (i + 1 < NK) {
                unsigned short* nxt = ((i + 1) & 1) ? lbuf1 : lbuf0;
                tdm_load_tile_64x32(Wtile + (i + 1) * 32,
                                    (unsigned)(uintptr_t)nxt);
                wait_tensorcnt<1>();   // tile i resident (in-order TDM)
            } else {
                wait_tensorcnt<0>();
            }
        }
#else
        stage_tile_fallback(Wtile + i * 32, cur, tid);
#endif
        __syncthreads();

        const int k0 = i * 32;
        const v16bf a0 = load_a(Abase0, k0, lane);
        const v16bf a1 = load_a(Abase1, k0, lane);
#pragma unroll
        for (int nf = 0; nf < 4; ++nf) {
            const v16bf b = load_b(cur + (nf * 16 + n16) * 32, lane);
            acc[0][nf] = wmma_bf16(a0, b, acc[0][nf]);
            acc[1][nf] = wmma_bf16(a1, b, acc[1][nf]);
        }
    }
}

// ---------------------------------------------------------------------------
// K0: convert x (fp32) -> bf16
// ---------------------------------------------------------------------------
__global__ __launch_bounds__(256) void cvt_x_kernel(const float* __restrict__ x,
                                                    unsigned short* __restrict__ xb,
                                                    int n) {
    int i = blockIdx.x * 256 + threadIdx.x;
    if (i < n) xb[i] = f2bf(x[i]);
}

// ---------------------------------------------------------------------------
// K1: transpose + convert weights: WT[z][n*1024 + k] = bf16(W[z][k*1024 + n])
// ---------------------------------------------------------------------------
__global__ __launch_bounds__(256) void cvt_wt_kernel(const float* __restrict__ W0,
                                                     const float* __restrict__ W1,
                                                     const float* __restrict__ W2,
                                                     const float* __restrict__ W3,
                                                     unsigned short* __restrict__ WT) {
    __shared__ float tile[32][33];
    const float* W = (blockIdx.z == 0) ? W0 :
                     (blockIdx.z == 1) ? W1 :
                     (blockIdx.z == 2) ? W2 : W3;
    unsigned short* dst = WT + (size_t)blockIdx.z * D_MODEL * D_MODEL;

    int x = blockIdx.x * 32 + threadIdx.x;
    int y = blockIdx.y * 32 + threadIdx.y;
#pragma unroll
    for (int j = 0; j < 32; j += 8)
        tile[threadIdx.y + j][threadIdx.x] = W[(size_t)(y + j) * D_MODEL + x];
    __syncthreads();
    int x2 = blockIdx.y * 32 + threadIdx.x;
    int y2 = blockIdx.x * 32 + threadIdx.y;
#pragma unroll
    for (int j = 0; j < 32; j += 8)
        dst[(size_t)(y2 + j) * D_MODEL + x2] = f2bf(tile[threadIdx.x][threadIdx.y + j]);
}

// ---------------------------------------------------------------------------
// K2: fused QKV projection GEMM.  grid = (N/64, M/128, 3), block = 128.
//   z=0: Q (scaled 1/8) -> [B,H,S,dk];  z=1: K -> [B,H,S,dk];
//   z=2: V (transposed) -> [B,H,dk,S]
// ---------------------------------------------------------------------------
__global__ __launch_bounds__(128) void gemm_qkv_kernel(const unsigned short* __restrict__ A,
                                                       const unsigned short* __restrict__ WT3,
                                                       unsigned short* __restrict__ Qb,
                                                       unsigned short* __restrict__ Kb,
                                                       unsigned short* __restrict__ Vt) {
    __shared__ __align__(16) unsigned short lbuf0[64 * 32];
    __shared__ __align__(16) unsigned short lbuf1[64 * 32];

    const int tid  = threadIdx.x;
    const int lane = tid & 31;
    const int wv   = tid >> 5;
    const int n16  = lane & 15;
    const int m0   = blockIdx.y * 128 + wv * 32;
    const int n0   = blockIdx.x * 64;
    const int z    = blockIdx.z;

    const unsigned short* WT = WT3 + (size_t)z * D_MODEL * D_MODEL;
    const unsigned short* Abase0 = A + (size_t)(m0 + n16) * D_MODEL;
    const unsigned short* Abase1 = A + (size_t)(m0 + 16 + n16) * D_MODEL;

    v8f acc[2][4] = {{{}, {}, {}, {}}, {{}, {}, {}, {}}};
    gemm_mainloop(Abase0, Abase1, WT + (size_t)n0 * D_MODEL,
                  lbuf0, lbuf1, lane, wv, tid, acc);

    const int half8 = (lane & 16) ? 8 : 0;
#pragma unroll
    for (int a = 0; a < 2; ++a) {
#pragma unroll
        for (int nf = 0; nf < 4; ++nf) {
            const int f = n0 + nf * 16 + n16;
            const int h = f >> 6, d = f & 63;
#pragma unroll
            for (int r = 0; r < 8; ++r) {
                const int M = m0 + a * 16 + r + half8;
                const int b = M >> 11, s = M & (SEQ - 1);
                const float v = acc[a][nf][r];
                if (z == 0) {
                    Qb[(((size_t)(b * NHEADS + h) * SEQ + s) << 6) + d] = f2bf(v * 0.125f);
                } else if (z == 1) {
                    Kb[(((size_t)(b * NHEADS + h) * SEQ + s) << 6) + d] = f2bf(v);
                } else {
                    Vt[((size_t)(b * NHEADS + h) * DKH + d) * SEQ + s] = f2bf(v);
                }
            }
        }
    }
}

// ---------------------------------------------------------------------------
// K3: causal flash attention.  grid = (B*H, S/64), block = 128 (4 waves).
// Each wave owns 16 query rows x full dk=64.  Q pre-scaled by 1/sqrt(dk).
// ---------------------------------------------------------------------------
__global__ __launch_bounds__(128) void attn_kernel(const unsigned short* __restrict__ Q,
                                                   const unsigned short* __restrict__ K,
                                                   const unsigned short* __restrict__ Vt,
                                                   unsigned short* __restrict__ AO) {
    __shared__ __align__(16) unsigned short Plds[4][16 * 64];

    const int bh    = blockIdx.x;
    const int q0    = blockIdx.y * 64;
    const int lane  = threadIdx.x & 31;
    const int wv    = threadIdx.x >> 5;
    const int n16   = lane & 15;
    const int half8 = (lane & 16) ? 8 : 0;

    const unsigned short* Qbase = Q + ((size_t)bh * SEQ + (q0 + wv * 16 + n16)) * DKH;
    const v16bf aQ0 = load_a(Qbase, 0, lane);
    const v16bf aQ1 = load_a(Qbase, 32, lane);

    v8f o0 = {}, o1 = {}, o2 = {}, o3 = {};
    float mrow[8], lrow[8];
#pragma unroll
    for (int r = 0; r < 8; ++r) { mrow[r] = NEG_INF; lrow[r] = 0.0f; }

    unsigned short* Pw = Plds[wv];
    const int nTiles = (q0 >> 6) + 1;

    for (int it = 0; it < nTiles; ++it) {
        const int t0 = it * 64;

        // scores: S = Q (16x64) @ K^T -> 4 fragments of 16x16
        v8f s0 = {}, s1 = {}, s2 = {}, s3 = {};
        {
            const unsigned short* Kb0 = K + ((size_t)bh * SEQ + (t0 +  0 + n16)) * DKH;
            const unsigned short* Kb1 = K + ((size_t)bh * SEQ + (t0 + 16 + n16)) * DKH;
            const unsigned short* Kb2 = K + ((size_t)bh * SEQ + (t0 + 32 + n16)) * DKH;
            const unsigned short* Kb3 = K + ((size_t)bh * SEQ + (t0 + 48 + n16)) * DKH;
            s0 = wmma_bf16(aQ0, load_b(Kb0, lane), s0);
            s0 = wmma_bf16(aQ1, load_b(Kb0 + 32, lane), s0);
            s1 = wmma_bf16(aQ0, load_b(Kb1, lane), s1);
            s1 = wmma_bf16(aQ1, load_b(Kb1 + 32, lane), s1);
            s2 = wmma_bf16(aQ0, load_b(Kb2, lane), s2);
            s2 = wmma_bf16(aQ1, load_b(Kb2 + 32, lane), s2);
            s3 = wmma_bf16(aQ0, load_b(Kb3, lane), s3);
            s3 = wmma_bf16(aQ1, load_b(Kb3 + 32, lane), s3);
        }

        // causal mask: only the diagonal tile can violate causality
        if (t0 == q0) {
#pragma unroll
            for (int r = 0; r < 8; ++r) {
                const int mG = q0 + wv * 16 + r + half8;
                if (t0 +  0 + n16 > mG) s0[r] = NEG_INF;
                if (t0 + 16 + n16 > mG) s1[r] = NEG_INF;
                if (t0 + 32 + n16 > mG) s2[r] = NEG_INF;
                if (t0 + 48 + n16 > mG) s3[r] = NEG_INF;
            }
        }

        // online softmax update (row reductions across 16-lane halves)
        float alpha[8];
#pragma unroll
        for (int r = 0; r < 8; ++r) {
            float tm = fmaxf(fmaxf(s0[r], s1[r]), fmaxf(s2[r], s3[r]));
            tm = fmaxf(tm, __shfl_xor(tm, 1));
            tm = fmaxf(tm, __shfl_xor(tm, 2));
            tm = fmaxf(tm, __shfl_xor(tm, 4));
            tm = fmaxf(tm, __shfl_xor(tm, 8));
            const float mnew = fmaxf(mrow[r], tm);
            alpha[r] = __expf(mrow[r] - mnew);
            mrow[r] = mnew;

            const float p0 = __expf(s0[r] - mnew);
            const float p1 = __expf(s1[r] - mnew);
            const float p2 = __expf(s2[r] - mnew);
            const float p3 = __expf(s3[r] - mnew);
            s0[r] = p0; s1[r] = p1; s2[r] = p2; s3[r] = p3;

            float rs = p0 + p1 + p2 + p3;
            rs += __shfl_xor(rs, 1);
            rs += __shfl_xor(rs, 2);
            rs += __shfl_xor(rs, 4);
            rs += __shfl_xor(rs, 8);
            lrow[r] = lrow[r] * alpha[r] + rs;

            o0[r] *= alpha[r]; o1[r] *= alpha[r];
            o2[r] *= alpha[r]; o3[r] *= alpha[r];
        }

        // P (C-layout fp32) -> LDS bf16 tile [16 x 64]
#pragma unroll
        for (int r = 0; r < 8; ++r) {
            const int row = (r + half8) * 64;
            Pw[row +  0 + n16] = f2bf(s0[r]);
            Pw[row + 16 + n16] = f2bf(s1[r]);
            Pw[row + 32 + n16] = f2bf(s2[r]);
            Pw[row + 48 + n16] = f2bf(s3[r]);
        }
        __syncthreads();

        // O += P (16x64) @ V (64x64); V stored transposed [dk][S]
        const unsigned short* Prow = Pw + (size_t)n16 * 64;
        const v16bf aP0 = load_a(Prow, 0, lane);
        const v16bf aP1 = load_a(Prow, 32, lane);

        const unsigned short* Vb0 = Vt + ((size_t)bh * DKH + ( 0 + n16)) * SEQ + t0;
        const unsigned short* Vb1 = Vt + ((size_t)bh * DKH + (16 + n16)) * SEQ + t0;
        const unsigned short* Vb2 = Vt + ((size_t)bh * DKH + (32 + n16)) * SEQ + t0;
        const unsigned short* Vb3 = Vt + ((size_t)bh * DKH + (48 + n16)) * SEQ + t0;
        o0 = wmma_bf16(aP0, load_b(Vb0, lane), o0);
        o0 = wmma_bf16(aP1, load_b(Vb0 + 32, lane), o0);
        o1 = wmma_bf16(aP0, load_b(Vb1, lane), o1);
        o1 = wmma_bf16(aP1, load_b(Vb1 + 32, lane), o1);
        o2 = wmma_bf16(aP0, load_b(Vb2, lane), o2);
        o2 = wmma_bf16(aP1, load_b(Vb2 + 32, lane), o2);
        o3 = wmma_bf16(aP0, load_b(Vb3, lane), o3);
        o3 = wmma_bf16(aP1, load_b(Vb3 + 32, lane), o3);
        __syncthreads();
    }

    // normalize and store to attn_out bf16 [B, S, E]
    const int b = bh >> 4, h = bh & 15;
#pragma unroll
    for (int r = 0; r < 8; ++r) {
        const int sG = q0 + wv * 16 + r + half8;
        const float inv = 1.0f / lrow[r];
        unsigned short* dst = AO + ((size_t)b * SEQ + sG) * D_MODEL + h * DKH + n16;
        dst[ 0] = f2bf(o0[r] * inv);
        dst[16] = f2bf(o1[r] * inv);
        dst[32] = f2bf(o2[r] * inv);
        dst[48] = f2bf(o3[r] * inv);
    }
}

// ---------------------------------------------------------------------------
// K4: output projection GEMM: out(fp32) = AO(bf16) @ WO (pre-transposed)
// grid = (N/64, M/128), block = 128
// ---------------------------------------------------------------------------
__global__ __launch_bounds__(128) void gemm_out_kernel(const unsigned short* __restrict__ A,
                                                       const unsigned short* __restrict__ WOT,
                                                       float* __restrict__ out) {
    __shared__ __align__(16) unsigned short lbuf0[64 * 32];
    __shared__ __align__(16) unsigned short lbuf1[64 * 32];

    const int tid  = threadIdx.x;
    const int lane = tid & 31;
    const int wv   = tid >> 5;
    const int n16  = lane & 15;
    const int m0   = blockIdx.y * 128 + wv * 32;
    const int n0   = blockIdx.x * 64;

    const unsigned short* Abase0 = A + (size_t)(m0 + n16) * D_MODEL;
    const unsigned short* Abase1 = A + (size_t)(m0 + 16 + n16) * D_MODEL;

    v8f acc[2][4] = {{{}, {}, {}, {}}, {{}, {}, {}, {}}};
    gemm_mainloop(Abase0, Abase1, WOT + (size_t)n0 * D_MODEL,
                  lbuf0, lbuf1, lane, wv, tid, acc);

    const int half8 = (lane & 16) ? 8 : 0;
#pragma unroll
    for (int a = 0; a < 2; ++a) {
#pragma unroll
        for (int r = 0; r < 8; ++r) {
            const size_t row = (size_t)(m0 + a * 16 + r + half8) * D_MODEL;
            out[row + n0 +  0 + n16] = acc[a][0][r];
            out[row + n0 + 16 + n16] = acc[a][1][r];
            out[row + n0 + 32 + n16] = acc[a][2][r];
            out[row + n0 + 48 + n16] = acc[a][3][r];
        }
    }
}

// ---------------------------------------------------------------------------
// host-side launcher
// ---------------------------------------------------------------------------
extern "C" void kernel_launch(void* const* d_in, const int* in_sizes, int n_in,
                              void* d_out, int out_size, void* d_ws, size_t ws_size,
                              hipStream_t stream) {
    (void)in_sizes; (void)n_in; (void)out_size; (void)ws_size;

    const float* x  = (const float*)d_in[0];
    const float* WQ = (const float*)d_in[1];
    const float* WK = (const float*)d_in[2];
    const float* WV = (const float*)d_in[3];
    const float* WO = (const float*)d_in[4];
    float* out = (float*)d_out;

    char* ws = (char*)d_ws;
    const size_t SZ_X   = (size_t)MTOT * D_MODEL * 2;
    const size_t SZ_W   = (size_t)D_MODEL * D_MODEL * 2;
    const size_t SZ_QKV = (size_t)BATCH * NHEADS * SEQ * DKH * 2;

    unsigned short* xb = (unsigned short*)(ws);
    unsigned short* WT = (unsigned short*)(ws + SZ_X);
    unsigned short* Qb = (unsigned short*)(ws + SZ_X + 4 * SZ_W);
    unsigned short* Kb = (unsigned short*)((char*)Qb + SZ_QKV);
    unsigned short* Vt = (unsigned short*)((char*)Kb + SZ_QKV);
    unsigned short* AO = (unsigned short*)((char*)Vt + SZ_QKV);

    {
        const int n = MTOT * D_MODEL;
        cvt_x_kernel<<<dim3(n / 256), dim3(256), 0, stream>>>(x, xb, n);
    }
    cvt_wt_kernel<<<dim3(32, 32, 4), dim3(32, 8), 0, stream>>>(WQ, WK, WV, WO, WT);

    gemm_qkv_kernel<<<dim3(D_MODEL / 64, MTOT / 128, 3), dim3(128), 0, stream>>>(
        xb, WT, Qb, Kb, Vt);

    attn_kernel<<<dim3(BATCH * NHEADS, SEQ / 64), dim3(128), 0, stream>>>(Qb, Kb, Vt, AO);

    gemm_out_kernel<<<dim3(D_MODEL / 64, MTOT / 128), dim3(128), 0, stream>>>(
        AO, WT + 3 * (size_t)D_MODEL * D_MODEL, out);
}
